// PairBiasedSelfAttention_36825049596581
// MI455X (gfx1250) — matrix operand order repaired and checked by reading, not compiled, not run on hardware
//
#include <hip/hip_runtime.h>
#include <hip/hip_bf16.h>

// ---------------- problem constants ----------------
#define B_    4
#define N_    512
#define D_    512
#define H_    16
#define DH_   32
#define PAIR_ 64
#define SCALE_ 0.17677669529663687f   // 1/sqrt(32)

typedef __attribute__((ext_vector_type(16))) __bf16 v16bf;
typedef __attribute__((ext_vector_type(8)))  __bf16 v8bf;
typedef __attribute__((ext_vector_type(4)))  __bf16 v4bf;
typedef __attribute__((ext_vector_type(8)))  float  v8f;

// ============== WMMA fragment loaders (contiguous k order, same in A and B; not the ISA table order) ==============
// A-matrix 16x32: lane holds row M=lane%16; the K pattern per lane is one
// contiguous 16-element run: [kofs+16*half .. +15].
template <typename T>
__device__ __forceinline__ v16bf load_a16(const T* row, int half, int kofs) {
    const T* p = row + kofs + 16 * half;
    const T* q = row + kofs + 16 * half + 8;
    v16bf a;
#pragma unroll
    for (int i = 0; i < 8; ++i) { a[i] = (__bf16)p[i]; a[8 + i] = (__bf16)q[i]; }
    return a;
}

// bf16 source: two aligned 16B vector loads.
__device__ __forceinline__ v16bf load_a16_bf(const __bf16* row, int half, int kofs) {
    v8bf lo = *(const v8bf*)(row + kofs + 16 * half);
    v8bf hi = *(const v8bf*)(row + kofs + 16 * half + 8);
    v16bf a;
#pragma unroll
    for (int i = 0; i < 8; ++i) { a[i] = lo[i]; a[8 + i] = hi[i]; }
    return a;
}

// B-matrix 32x16 (KxN) where memory holds the TRANSPOSE row-major
// (mat[col][k], row stride ldm): fragment = 16 contiguous bf16 -> one 32B load.
// b[e] = B[k = kofs + 16*half + e][col]
__device__ __forceinline__ v16bf load_bT_bf(const __bf16* mat, int ldm, int col,
                                            int half, int kofs) {
    return *(const v16bf*)(mat + (size_t)col * ldm + kofs + 16 * half);
}

// Pre-packed B fragments: [ntile][kt][lane][16] -> one 32B load per lane.
__device__ __forceinline__ v16bf load_bpack(const __bf16* base, int ntile, int nkt,
                                            int kt, int lane) {
    return *(const v16bf*)(base + ((size_t)(ntile * nkt + kt) * 32 + lane) * 16);
}

__device__ __forceinline__ v8f wmma_bf16(v16bf a, v16bf b, v8f c) {
    return __builtin_amdgcn_wmma_f32_16x16x32_bf16(false, a, false, b, (short)0, c,
                                                   false, false);
}

// =====================================================================
// Prep A: pack an f32 matrix [rows, cols] into bf16 B-fragment layout.
// =====================================================================
__global__ void pack_b_kernel(const float* __restrict__ W, __bf16* __restrict__ out,
                              int rows, int cols) {
    int idx = blockIdx.x * blockDim.x + threadIdx.x;
    if (idx >= rows * cols) return;
    int k = idx / cols, col = idx % cols;
    int ntile = col >> 4, c = col & 15;
    int kt = k >> 5, krem = k & 31;
    int half = krem >> 4, e = krem & 15;
    int lane = half * 16 + c;
    int nkt = rows >> 5;
    out[((size_t)(ntile * nkt + kt) * 32 + lane) * 16 + e] = (__bf16)W[idx];
}

// Prep B: f32 -> bf16 elementwise (x conversion), 4 elements per thread.
__global__ void cvt_bf16_kernel(const float* __restrict__ src,
                                __bf16* __restrict__ dst, int n4) {
    int i = blockIdx.x * blockDim.x + threadIdx.x;
    if (i >= n4) return;
    float4 v = ((const float4*)src)[i];
    v4bf o = { (__bf16)v.x, (__bf16)v.y, (__bf16)v.z, (__bf16)v.w };
    ((v4bf*)dst)[i] = o;
}

// =====================================================================
// Kernel 1: QKV projection. blockIdx.y selects Wq/Wk/Wv (wave-uniform).
// Each wave computes a 16x64 output strip (4 N-tiles): A fragment loaded
// once per K-step, 4 WMMAs. 1024 waves per matrix, grid (128, 3) x 256.
// =====================================================================
__global__ void qkv_kernel(const __bf16* __restrict__ xb,
                           const __bf16* __restrict__ WqP,
                           const __bf16* __restrict__ WkP,
                           const __bf16* __restrict__ WvP,
                           __bf16* __restrict__ Qb,
                           __bf16* __restrict__ Kb,
                           __bf16* __restrict__ Vt) {
    const int which = blockIdx.y;                 // 0=Q, 1=K, 2=V (uniform)
    const __bf16* WP = (which == 0) ? WqP : (which == 1) ? WkP : WvP;

    const int lane = threadIdx.x & 31;
    const int wid  = blockIdx.x * (blockDim.x >> 5) + (threadIdx.x >> 5);
    const int mtile  = wid >> 3;                  // 128 m-tiles
    const int ntile0 = (wid & 7) * 4;             // 8 n-blocks of 4 tiles

    const int rr   = lane & 15;
    const int half = lane >> 4;

    const __bf16* arow = xb + (size_t)(mtile * 16 + rr) * D_;
    v8f acc[4] = {{}, {}, {}, {}};
#pragma unroll 4
    for (int kt = 0; kt < D_ / 32; ++kt) {
        v16bf a = load_a16_bf(arow, half, kt * 32);
#pragma unroll
        for (int t = 0; t < 4; ++t) {
            v16bf bf = load_bpack(WP, ntile0 + t, D_ / 32, kt, lane);
            acc[t] = wmma_bf16(a, bf, acc[t]);
        }
    }
#pragma unroll
    for (int t = 0; t < 4; ++t) {
#pragma unroll
        for (int r = 0; r < 8; ++r) {
            int m   = mtile * 16 + r + 8 * half;         // global row in [B*N]
            int col = (ntile0 + t) * 16 + rr;            // global col in [D]
            int b = m >> 9, n = m & (N_ - 1);
            int h = col >> 5, dh = col & (DH_ - 1);
            __bf16 v = (__bf16)acc[t][r];
            if (which == 2) {
                Vt[((size_t)(b * H_ + h) * DH_ + dh) * N_ + n] = v;   // [B,H,DH,N]
            } else if (which == 0) {
                Qb[((size_t)(b * H_ + h) * N_ + n) * DH_ + dh] = v;   // [B,H,N,DH]
            } else {
                Kb[((size_t)(b * H_ + h) * N_ + n) * DH_ + dh] = v;
            }
        }
    }
}

// =====================================================================
// Kernel 2: bias projection  pair_bias[B*N*N, 64] x WpPack -> d_out attn
// region [B,H,N,N] (bias scratch), pair-masked.  65536 waves.
// =====================================================================
__global__ void bias_kernel(const float* __restrict__ pb,
                            const __bf16* __restrict__ WpP,
                            const unsigned char* __restrict__ mask,
                            float* __restrict__ biasOut) {
    const int lane = threadIdx.x & 31;
    const int wid  = blockIdx.x * (blockDim.x >> 5) + (threadIdx.x >> 5);
    const long base = (long)wid * 16;            // first flat (b,n,m) row
    const int rr   = lane & 15;
    const int half = lane >> 4;

    const float* arow = pb + (size_t)(base + rr) * PAIR_;
    __builtin_prefetch(arow + 16 * PAIR_, 0, 1);   // next row-block of the stream
    v8f acc = {};
#pragma unroll
    for (int kt = 0; kt < PAIR_ / 32; ++kt) {
        v16bf a  = load_a16(arow, half, kt * 32);
        v16bf bf = load_bpack(WpP, 0, PAIR_ / 32, kt, lane);
        acc = wmma_bf16(a, bf, acc);
    }
    const int h = rr;
#pragma unroll
    for (int r = 0; r < 8; ++r) {
        long flat = base + r + 8 * half;         // flat (b,n,m)
        int b  = (int)(flat >> 18);              // / (N*N)
        int nm = (int)(flat & ((1 << 18) - 1));
        int n = nm >> 9, m = nm & (N_ - 1);
        float v = acc[r];
        if (!(mask[b * N_ + n] && mask[b * N_ + m])) v = 0.0f;
        biasOut[((size_t)(b * H_ + h) * N_ + n) * N_ + m] = v;
    }
}

// =====================================================================
// Kernel 3: attention.  One WG (256 thr = 8 waves) per (b,h,ntile).
// =====================================================================
__global__ void attn_kernel(const __bf16* __restrict__ Qb,
                            const __bf16* __restrict__ Kb,
                            const __bf16* __restrict__ Vt,
                            const unsigned char* __restrict__ mask,
                            float* __restrict__ attnB,   // bias in, attn out
                            __bf16* __restrict__ Ob) {
    __shared__ float panel[16][N_];
    __shared__ float red[16][16];
    __shared__ float rowstat[16];

    const int bh    = blockIdx.x >> 5;           // 32 n-tiles
    const int ntile = blockIdx.x & 31;
    const int b     = bh >> 4;
    const int tid   = threadIdx.x;
    const int wv    = tid >> 5;
    const int lane  = tid & 31;
    const int rr    = lane & 15;
    const int half  = lane >> 4;

    const __bf16* Qbase  = Qb + (size_t)bh * N_ * DH_;
    const __bf16* Kbase  = Kb + (size_t)bh * N_ * DH_;
    const __bf16* Vtbase = Vt + (size_t)bh * DH_ * N_;
    float* biasRow = attnB + ((size_t)bh * N_ + ntile * 16) * N_;

    // ---- phase 1: score panel (1 WMMA per 16x16 tile; K = DH = 32) ----
    v16bf qf = load_a16_bf(Qbase + (size_t)(ntile * 16 + rr) * DH_, half, 0);
    for (int mt = wv; mt < N_ / 16; mt += 8) {
        v16bf kf = load_bT_bf(Kbase + (size_t)mt * 16 * DH_, DH_, rr, half, 0);
        v8f acc = {};
        acc = wmma_bf16(qf, kf, acc);
#pragma unroll
        for (int r = 0; r < 8; ++r) {
            int nl = r + 8 * half;               // local row
            int m  = mt * 16 + rr;               // global key index
            float v = acc[r] * SCALE_ + biasRow[(size_t)nl * N_ + m];
            if (!mask[b * N_ + m]) v = -1.0e6f;
            panel[nl][m] = v;
        }
    }
    __syncthreads();

    // ---- phase 2: exact softmax over each of 16 rows (16 threads/row) ----
    {
        const int row = tid >> 4;
        const int c   = tid & 15;
        float mx = -3.0e38f;
        for (int j = c; j < N_; j += 16) mx = fmaxf(mx, panel[row][j]);
        red[row][c] = mx;
        __syncthreads();
        if (c == 0) {
            float m2 = red[row][0];
#pragma unroll
            for (int t = 1; t < 16; ++t) m2 = fmaxf(m2, red[row][t]);
            rowstat[row] = m2;
        }
        __syncthreads();
        float rm = rowstat[row];
        float s = 0.0f;
        for (int j = c; j < N_; j += 16) {
            float e = __expf(panel[row][j] - rm);
            panel[row][j] = e;
            s += e;
        }
        red[row][c] = s;
        __syncthreads();
        if (c == 0) {
            float t2 = 0.0f;
#pragma unroll
            for (int t = 0; t < 16; ++t) t2 += red[row][t];
            rowstat[row] = 1.0f / t2;
        }
        __syncthreads();
        float inv = rowstat[row];
        float* orow = biasRow + (size_t)row * N_;   // overwrite bias with attn
        for (int j = c; j < N_; j += 16) {
            float p = panel[row][j] * inv;
            panel[row][j] = p;
            orow[j] = p;
        }
    }
    __syncthreads();

    // ---- phase 3: O tile = attn(16x512) x V(512x32); waves 0,1 take the
    //      two 16-col halves of DH; V fragments are contiguous in Vt ----
    if (wv < 2) {
        v8f acc = {};
        for (int kt = 0; kt < N_ / 32; ++kt) {
            v16bf a  = load_a16(&panel[rr][0], half, kt * 32);
            v16bf bf = load_bT_bf(Vtbase, N_, wv * 16 + rr, half, kt * 32);
            acc = wmma_bf16(a, bf, acc);
        }
        const int h = bh & 15;
#pragma unroll
        for (int r = 0; r < 8; ++r) {
            int n  = ntile * 16 + r + 8 * half;
            int dh = wv * 16 + rr;
            Ob[(size_t)(b * N_ + n) * D_ + h * DH_ + dh] = (__bf16)acc[r];
        }
    }
}

// =====================================================================
// Kernel 4: final = O x Wo, rows zeroed where !mask.  Each wave does a
// 16x64 strip (4 N-tiles, shared A fragment).  1024 waves.
// =====================================================================
__global__ void final_kernel(const __bf16* __restrict__ Ob,
                             const __bf16* __restrict__ WoP,
                             const unsigned char* __restrict__ mask,
                             float* __restrict__ out) {
    const int lane = threadIdx.x & 31;
    const int wid  = blockIdx.x * (blockDim.x >> 5) + (threadIdx.x >> 5);
    const int mtile  = wid >> 3;
    const int ntile0 = (wid & 7) * 4;
    const int rr   = lane & 15;
    const int half = lane >> 4;

    const __bf16* arow = Ob + (size_t)(mtile * 16 + rr) * D_;
    v8f acc[4] = {{}, {}, {}, {}};
#pragma unroll 4
    for (int kt = 0; kt < D_ / 32; ++kt) {
        v16bf a = load_a16_bf(arow, half, kt * 32);
#pragma unroll
        for (int t = 0; t < 4; ++t) {
            v16bf bf = load_bpack(WoP, ntile0 + t, D_ / 32, kt, lane);
            acc[t] = wmma_bf16(a, bf, acc[t]);
        }
    }
#pragma unroll
    for (int t = 0; t < 4; ++t) {
#pragma unroll
        for (int r = 0; r < 8; ++r) {
            int m   = mtile * 16 + r + 8 * half;
            int col = (ntile0 + t) * 16 + rr;
            int b = m >> 9, n = m & (N_ - 1);
            float v = mask[b * N_ + n] ? acc[t][r] : 0.0f;
            out[(size_t)m * D_ + col] = v;
        }
    }
}

// =====================================================================
extern "C" void kernel_launch(void* const* d_in, const int* in_sizes, int n_in,
                              void* d_out, int out_size, void* d_ws, size_t ws_size,
                              hipStream_t stream) {
    (void)in_sizes; (void)n_in; (void)out_size; (void)ws_size;

    const float* x    = (const float*)d_in[0];
    const float* pb   = (const float*)d_in[1];
    const float* Wq   = (const float*)d_in[2];
    const float* Wk   = (const float*)d_in[3];
    const float* Wv   = (const float*)d_in[4];
    const float* Wo   = (const float*)d_in[5];
    const float* Wp   = (const float*)d_in[6];
    const unsigned char* mask = (const unsigned char*)d_in[7];

    float* finalOut = (float*)d_out;                                  // [B,N,D]
    float* attnOut  = finalOut + (size_t)B_ * N_ * D_;                // [B,H,N,N]

    // ---- workspace layout (bf16 elements), ~12 MB total ----
    const size_t E = (size_t)B_ * N_ * D_;                            // 1,048,576
    __bf16* xb  = (__bf16*)d_ws;        // [B*N, D]
    __bf16* Qb  = xb  + E;              // [B,H,N,DH]
    __bf16* Kb  = Qb  + E;              // [B,H,N,DH]
    __bf16* Vt  = Kb  + E;              // [B,H,DH,N]
    __bf16* Ob  = Vt  + E;              // [B,N,D]
    __bf16* WqP = Ob  + E;              // packed fragments, D*D each
    __bf16* WkP = WqP + (size_t)D_ * D_;
    __bf16* WvP = WkP + (size_t)D_ * D_;
    __bf16* WoP = WvP + (size_t)D_ * D_;
    __bf16* WpP = WoP + (size_t)D_ * D_;   // PAIR_*H_ = 1024 elements

    // ---- prep: pack weights to fragment layout; convert x to bf16 ----
    pack_b_kernel<<<dim3((D_ * D_ + 255) / 256), dim3(256), 0, stream>>>(Wq, WqP, D_, D_);
    pack_b_kernel<<<dim3((D_ * D_ + 255) / 256), dim3(256), 0, stream>>>(Wk, WkP, D_, D_);
    pack_b_kernel<<<dim3((D_ * D_ + 255) / 256), dim3(256), 0, stream>>>(Wv, WvP, D_, D_);
    pack_b_kernel<<<dim3((D_ * D_ + 255) / 256), dim3(256), 0, stream>>>(Wo, WoP, D_, D_);
    pack_b_kernel<<<dim3((PAIR_ * H_ + 255) / 256), dim3(256), 0, stream>>>(Wp, WpP, PAIR_, H_);
    cvt_bf16_kernel<<<dim3((int)(E / 4 / 256)), dim3(256), 0, stream>>>(x, xb, (int)(E / 4));

    // ---- main pipeline ----
    qkv_kernel<<<dim3(128, 3), dim3(256), 0, stream>>>(xb, WqP, WkP, WvP, Qb, Kb, Vt);
    bias_kernel<<<dim3(8192), dim3(256), 0, stream>>>(pb, WpP, mask, attnOut);
    attn_kernel<<<dim3(B_ * H_ * (N_ / 16)), dim3(256), 0, stream>>>(
        Qb, Kb, Vt, mask, attnOut, Ob);
    final_kernel<<<dim3(128), dim3(256), 0, stream>>>(Ob, WoP, mask, finalOut);
}
